// rnn_crf_58110907515188
// MI455X (gfx1250) — matrix operand order repaired and checked by reading, not compiled, not run on hardware
//
#include <hip/hip_runtime.h>
#include <math.h>

typedef __attribute__((ext_vector_type(16))) _Float16 v16h;
typedef __attribute__((ext_vector_type(8)))  _Float16 v8h;
typedef __attribute__((ext_vector_type(8)))  float    v8f;

#define T_LEN  256
#define B_SZ   64
#define E_DIM  512
#define H_DIM  512
#define G_DIM  2048   // 4*H
#define K_TAGS 24
#define NEGV   (-10000.0f)
#define LDH    520    // padded LDS row stride (halves) for h
#define NW     4      // cooperating workgroups per direction
#define HW     (H_DIM / NW)   // 128 h-columns per workgroup

// ---------- WMMA helpers (A and B share contiguous k order k = 16*(lane/16) + s) ----------

__device__ __forceinline__ v8f wmma_f32(v16h a, v16h b, v8f c) {
  return __builtin_amdgcn_wmma_f32_16x16x32_f16(false, a, false, b, (short)0, c,
                                                false, false);
}

// A fragment: 16x32 f16, row-major source with row stride lda (elements).
// lane<16: row=lane, K = {0..15}; lane>=16: row=lane-16, K={16..31}
__device__ __forceinline__ v16h load_a_frag(const _Float16* A, int lda, int lane) {
  int row  = lane & 15;
  int koff = (lane >> 4) << 4;
  const _Float16* p = A + (size_t)row * lda + koff;
  v8h lo = *(const v8h*)(p);
  v8h hi = *(const v8h*)(p + 8);
  v16h a;
#pragma unroll
  for (int i = 0; i < 8; ++i) { a[i] = lo[i]; a[i + 8] = hi[i]; }
  return a;
}

// B fragment: 32x16 (KxN) where B[k,n] = W[n,k], W row-major (N rows x K cols),
// caller pre-offsets W to (n0,k0). lane: n = lane&15, k-half = lane>>4,
// 16 contiguous k values per lane -> single 32-byte load.
__device__ __forceinline__ v16h load_b_frag(const _Float16* W, int ldw, int lane) {
  const _Float16* p = W + (size_t)(lane & 15) * ldw + ((lane >> 4) << 4);
  return *(const v16h*)p;
}

// ---------- small prep kernels ----------

__global__ void cvt_f16(const float* __restrict__ src, _Float16* __restrict__ dst, int n) {
  int i = blockIdx.x * blockDim.x + threadIdx.x;
  if (i < n) dst[i] = (_Float16)src[i];
}

__global__ void pad_wout(const float* __restrict__ w, _Float16* __restrict__ dst) {
  // dst: 32 x 1024 f16, rows >= 24 zero
  int i = blockIdx.x * blockDim.x + threadIdx.x;   // 32768
  int row = i >> 10, col = i & 1023;
  dst[i] = (row < K_TAGS) ? (_Float16)w[row * (2 * H_DIM) + col] : (_Float16)0.0f;
}

__global__ void gather_embed(const int* __restrict__ xw, const float* __restrict__ emb,
                             _Float16* __restrict__ x16) {
  int i = blockIdx.x * blockDim.x + threadIdx.x;   // T*B*E
  int row = i >> 9;                                // t*B+b
  int e   = i & 511;
  int tok = xw[row];
  x16[i] = (_Float16)emb[(size_t)tok * E_DIM + e];
}

__global__ void init_state(_Float16* __restrict__ hg, unsigned* __restrict__ barcnt) {
  int i = blockIdx.x * blockDim.x + threadIdx.x;
  if (i < 2 * B_SZ * H_DIM) hg[i] = (_Float16)0.0f;
  if (i < 128) barcnt[i] = 0u;
}

// ---------- xg = x @ w_ih^T + b : M=16384, N=2048, K=512, out f16 ----------
// Software-pipelined: fragments for k+32 are in flight while WMMAs consume k.

__global__ void gemm_xg(const _Float16* __restrict__ A, const _Float16* __restrict__ W,
                        const float* __restrict__ bias, _Float16* __restrict__ out) {
  int wid  = (blockIdx.x * blockDim.x + threadIdx.x) >> 5;
  int lane = threadIdx.x & 31;
  int ngrp = wid & 31;           // 32 groups of 64 output cols
  int mt   = wid >> 5;           // 1024 row tiles
  if (mt >= (T_LEN * B_SZ) / 16) return;   // wave-uniform
  int n0 = ngrp * 64;
  const _Float16* Ab = A + (size_t)mt * 16 * E_DIM;

  v8f acc[4];
#pragma unroll
  for (int nt = 0; nt < 4; ++nt)
#pragma unroll
    for (int j = 0; j < 8; ++j) acc[nt][j] = 0.0f;

  v16h a = load_a_frag(Ab, E_DIM, lane);
  v16h b0 = load_b_frag(W + (size_t)(n0 +  0) * E_DIM, E_DIM, lane);
  v16h b1 = load_b_frag(W + (size_t)(n0 + 16) * E_DIM, E_DIM, lane);
  v16h b2 = load_b_frag(W + (size_t)(n0 + 32) * E_DIM, E_DIM, lane);
  v16h b3 = load_b_frag(W + (size_t)(n0 + 48) * E_DIM, E_DIM, lane);

#pragma unroll 1
  for (int k0 = 32; k0 < E_DIM; k0 += 32) {
    v16h an  = load_a_frag(Ab + k0, E_DIM, lane);
    v16h bn0 = load_b_frag(W + (size_t)(n0 +  0) * E_DIM + k0, E_DIM, lane);
    v16h bn1 = load_b_frag(W + (size_t)(n0 + 16) * E_DIM + k0, E_DIM, lane);
    v16h bn2 = load_b_frag(W + (size_t)(n0 + 32) * E_DIM + k0, E_DIM, lane);
    v16h bn3 = load_b_frag(W + (size_t)(n0 + 48) * E_DIM + k0, E_DIM, lane);
    acc[0] = wmma_f32(a, b0, acc[0]);
    acc[1] = wmma_f32(a, b1, acc[1]);
    acc[2] = wmma_f32(a, b2, acc[2]);
    acc[3] = wmma_f32(a, b3, acc[3]);
    a = an; b0 = bn0; b1 = bn1; b2 = bn2; b3 = bn3;
  }
  acc[0] = wmma_f32(a, b0, acc[0]);
  acc[1] = wmma_f32(a, b1, acc[1]);
  acc[2] = wmma_f32(a, b2, acc[2]);
  acc[3] = wmma_f32(a, b3, acc[3]);

  int rb = mt * 16 + ((lane >> 4) << 3);
  int cc = lane & 15;
#pragma unroll
  for (int nt = 0; nt < 4; ++nt) {
    int col = n0 + nt * 16 + cc;
    float bv = bias[col];
#pragma unroll
    for (int j = 0; j < 8; ++j)
      out[(size_t)(rb + j) * G_DIM + col] = (_Float16)(acc[nt][j] + bv);
  }
}

// ---------- multi-workgroup persistent LSTM --------------------------------
// 2*NW workgroups (NW per direction), 256 threads = 8 waves each.
// Wave w owns h-column tile j0 = slot*HW + w*16 : computes i,f,g,o gates for
// those 16 columns over all 64 batch rows (4 M-tiles x 4 gates = 16 WMMA
// accumulators). c-state stays in VGPRs in the WMMA C/D layout. h is
// exchanged each step through a global f16 buffer; the broadcast back into
// LDS uses global_load_async_to_lds_b128 (ASYNCcnt).

__global__ void lstm_multi(const _Float16* __restrict__ xg_f,
                           const _Float16* __restrict__ xg_b,
                           const _Float16* __restrict__ whh_f,
                           const _Float16* __restrict__ whh_b,
                           _Float16* __restrict__ hf, _Float16* __restrict__ hb,
                           _Float16* __restrict__ hg_all,
                           unsigned* __restrict__ barcnt) {
  __shared__ __align__(32) _Float16 hsh[B_SZ * LDH];   // 66.6 KB of 320 KB/WGP
  const int wg   = blockIdx.x;            // 0..2*NW-1
  const int rev  = (wg >= NW) ? 1 : 0;
  const int slot = rev ? (wg - NW) : wg;  // 0..NW-1
  const _Float16* xg  = rev ? xg_b  : xg_f;
  const _Float16* whh = rev ? whh_b : whh_f;
  _Float16* hout      = rev ? hb : hf;
  _Float16* hg        = hg_all + (size_t)rev * B_SZ * H_DIM;
  unsigned* cnt       = barcnt + (rev ? 64 : 0);

  const int tid  = threadIdx.x;      // 0..255
  const int wave = tid >> 5;         // 0..7
  const int lane = tid & 31;
  const int rb   = (lane >> 4) << 3; // 0 or 8
  const int cc   = lane & 15;
  const int j0   = slot * HW + wave * 16;   // this wave's 16 h-columns
  const unsigned lds_base = (unsigned)(size_t)(&hsh[0]);

  float cst[4][8];
#pragma unroll
  for (int mt = 0; mt < 4; ++mt)
#pragma unroll
    for (int j = 0; j < 8; ++j) cst[mt][j] = 0.0f;

  for (int ts = 0; ts < T_LEN; ++ts) {
    const int t = rev ? (T_LEN - 1 - ts) : ts;

    // ---- broadcast h(step) from global into LDS via async copy ----
    // 64 rows x 512 halves = 4096 x 16B chunks, 256 threads x 16 iters.
#pragma unroll 1
    for (int it = 0; it < 16; ++it) {
      int chunk = it * 256 + tid;
      int r = chunk >> 6;
      int c = chunk & 63;
      unsigned lofs = lds_base + (unsigned)(r * (LDH * 2) + c * 16);
      unsigned long long gaddr =
          (unsigned long long)(const void*)hg + (unsigned long long)(r * 1024 + c * 16);
      asm volatile("global_load_async_to_lds_b128 %0, %1, off"
                   :: "v"(lofs), "v"(gaddr) : "memory");
    }
    asm volatile("s_wait_asynccnt 0x0" ::: "memory");
    __syncthreads();

    // ---- accumulators start at the precomputed xg gate pre-activations ----
    const _Float16* xgt = xg + (size_t)t * B_SZ * G_DIM;
    v8f acc[4][4];
#pragma unroll
    for (int mt = 0; mt < 4; ++mt)
#pragma unroll
      for (int gi = 0; gi < 4; ++gi) {
        int col = gi * H_DIM + j0 + cc;
#pragma unroll
        for (int j = 0; j < 8; ++j)
          acc[mt][gi][j] = (float)xgt[(size_t)(mt * 16 + rb + j) * G_DIM + col];
      }

    // ---- gates += h @ w_hh^T (K=512), pipelined w_hh fragment loads ----
    v16h bc0 = load_b_frag(whh + (size_t)(0 * H_DIM + j0) * H_DIM, H_DIM, lane);
    v16h bc1 = load_b_frag(whh + (size_t)(1 * H_DIM + j0) * H_DIM, H_DIM, lane);
    v16h bc2 = load_b_frag(whh + (size_t)(2 * H_DIM + j0) * H_DIM, H_DIM, lane);
    v16h bc3 = load_b_frag(whh + (size_t)(3 * H_DIM + j0) * H_DIM, H_DIM, lane);

#pragma unroll 1
    for (int k0 = 0; k0 < H_DIM - 32; k0 += 32) {
      v16h a0 = load_a_frag(&hsh[0 * 16 * LDH] + k0, LDH, lane);
      v16h a1 = load_a_frag(&hsh[1 * 16 * LDH] + k0, LDH, lane);
      v16h a2 = load_a_frag(&hsh[2 * 16 * LDH] + k0, LDH, lane);
      v16h a3 = load_a_frag(&hsh[3 * 16 * LDH] + k0, LDH, lane);
      int k1 = k0 + 32;
      v16h bn0 = load_b_frag(whh + (size_t)(0 * H_DIM + j0) * H_DIM + k1, H_DIM, lane);
      v16h bn1 = load_b_frag(whh + (size_t)(1 * H_DIM + j0) * H_DIM + k1, H_DIM, lane);
      v16h bn2 = load_b_frag(whh + (size_t)(2 * H_DIM + j0) * H_DIM + k1, H_DIM, lane);
      v16h bn3 = load_b_frag(whh + (size_t)(3 * H_DIM + j0) * H_DIM + k1, H_DIM, lane);
      acc[0][0] = wmma_f32(a0, bc0, acc[0][0]);
      acc[1][0] = wmma_f32(a1, bc0, acc[1][0]);
      acc[2][0] = wmma_f32(a2, bc0, acc[2][0]);
      acc[3][0] = wmma_f32(a3, bc0, acc[3][0]);
      acc[0][1] = wmma_f32(a0, bc1, acc[0][1]);
      acc[1][1] = wmma_f32(a1, bc1, acc[1][1]);
      acc[2][1] = wmma_f32(a2, bc1, acc[2][1]);
      acc[3][1] = wmma_f32(a3, bc1, acc[3][1]);
      acc[0][2] = wmma_f32(a0, bc2, acc[0][2]);
      acc[1][2] = wmma_f32(a1, bc2, acc[1][2]);
      acc[2][2] = wmma_f32(a2, bc2, acc[2][2]);
      acc[3][2] = wmma_f32(a3, bc2, acc[3][2]);
      acc[0][3] = wmma_f32(a0, bc3, acc[0][3]);
      acc[1][3] = wmma_f32(a1, bc3, acc[1][3]);
      acc[2][3] = wmma_f32(a2, bc3, acc[2][3]);
      acc[3][3] = wmma_f32(a3, bc3, acc[3][3]);
      bc0 = bn0; bc1 = bn1; bc2 = bn2; bc3 = bn3;
    }
    {
      const int k0 = H_DIM - 32;
      v16h a0 = load_a_frag(&hsh[0 * 16 * LDH] + k0, LDH, lane);
      v16h a1 = load_a_frag(&hsh[1 * 16 * LDH] + k0, LDH, lane);
      v16h a2 = load_a_frag(&hsh[2 * 16 * LDH] + k0, LDH, lane);
      v16h a3 = load_a_frag(&hsh[3 * 16 * LDH] + k0, LDH, lane);
      acc[0][0] = wmma_f32(a0, bc0, acc[0][0]);
      acc[1][0] = wmma_f32(a1, bc0, acc[1][0]);
      acc[2][0] = wmma_f32(a2, bc0, acc[2][0]);
      acc[3][0] = wmma_f32(a3, bc0, acc[3][0]);
      acc[0][1] = wmma_f32(a0, bc1, acc[0][1]);
      acc[1][1] = wmma_f32(a1, bc1, acc[1][1]);
      acc[2][1] = wmma_f32(a2, bc1, acc[2][1]);
      acc[3][1] = wmma_f32(a3, bc1, acc[3][1]);
      acc[0][2] = wmma_f32(a0, bc2, acc[0][2]);
      acc[1][2] = wmma_f32(a1, bc2, acc[1][2]);
      acc[2][2] = wmma_f32(a2, bc2, acc[2][2]);
      acc[3][2] = wmma_f32(a3, bc2, acc[3][2]);
      acc[0][3] = wmma_f32(a0, bc3, acc[0][3]);
      acc[1][3] = wmma_f32(a1, bc3, acc[1][3]);
      acc[2][3] = wmma_f32(a2, bc3, acc[2][3]);
      acc[3][3] = wmma_f32(a3, bc3, acc[3][3]);
    }

    // ---- LSTM cell for this wave's 16 columns (i,f,g,o order) ----
#pragma unroll
    for (int mt = 0; mt < 4; ++mt)
#pragma unroll
      for (int j = 0; j < 8; ++j) {
        float ig = acc[mt][0][j], fg = acc[mt][1][j];
        float gg = acc[mt][2][j], og = acc[mt][3][j];
        float si = 1.0f / (1.0f + __expf(-ig));
        float sf = 1.0f / (1.0f + __expf(-fg));
        float so = 1.0f / (1.0f + __expf(-og));
        float cn = sf * cst[mt][j] + si * tanhf(gg);
        float hn = so * tanhf(cn);
        cst[mt][j] = cn;
        int r   = mt * 16 + rb + j;
        int col = j0 + cc;
        _Float16 h16 = (_Float16)hn;
        hg[(size_t)r * H_DIM + col] = h16;
        hout[((size_t)t * B_SZ + r) * H_DIM + col] = h16;
      }

    // ---- device-scope step barrier across the NW workgroups ----
    __threadfence();          // release our hg stores to device scope
    __syncthreads();
    if (tid == 0) {
      atomicAdd(cnt, 1u);
      const unsigned tgt = (unsigned)(ts + 1) * NW;
      while (__hip_atomic_load(cnt, __ATOMIC_ACQUIRE, __HIP_MEMORY_SCOPE_AGENT) < tgt) {
        __builtin_amdgcn_s_sleep(1);
      }
    }
    __syncthreads();
    __threadfence();          // acquire: invalidate before re-reading hg
  }
}

// ---------- projection: [hf|hb](16384x1024) @ w_out^T(24x1024) + b, *mask ----

__global__ void gemm_proj(const _Float16* __restrict__ hfp, const _Float16* __restrict__ hbp,
                          const _Float16* __restrict__ wout16, const float* __restrict__ bout,
                          const int* __restrict__ y0, float* __restrict__ hp) {
  int wid  = (blockIdx.x * blockDim.x + threadIdx.x) >> 5;
  int lane = threadIdx.x & 31;
  if (wid >= (T_LEN * B_SZ) / 16) return;   // wave-uniform
  int mt = wid;

  v8f acc[2];
#pragma unroll
  for (int nt = 0; nt < 2; ++nt)
#pragma unroll
    for (int j = 0; j < 8; ++j) acc[nt][j] = 0.0f;

  for (int k0 = 0; k0 < H_DIM; k0 += 32) {           // forward half of concat
    v16h a = load_a_frag(hfp + (size_t)mt * 16 * H_DIM + k0, H_DIM, lane);
#pragma unroll
    for (int nt = 0; nt < 2; ++nt) {
      v16h b = load_b_frag(wout16 + (size_t)(nt * 16) * (2 * H_DIM) + k0,
                           2 * H_DIM, lane);
      acc[nt] = wmma_f32(a, b, acc[nt]);
    }
  }
  for (int k0 = 0; k0 < H_DIM; k0 += 32) {           // backward half of concat
    v16h a = load_a_frag(hbp + (size_t)mt * 16 * H_DIM + k0, H_DIM, lane);
#pragma unroll
    for (int nt = 0; nt < 2; ++nt) {
      v16h b = load_b_frag(wout16 + (size_t)(nt * 16) * (2 * H_DIM) + H_DIM + k0,
                           2 * H_DIM, lane);
      acc[nt] = wmma_f32(a, b, acc[nt]);
    }
  }
  int rb = mt * 16 + ((lane >> 4) << 3);
  int cc = lane & 15;
#pragma unroll
  for (int nt = 0; nt < 2; ++nt) {
    int col = nt * 16 + cc;
    if (col < K_TAGS) {
      float bv = bout[col];
#pragma unroll
      for (int j = 0; j < 8; ++j) {
        int r = rb + j;                             // r = t*B + b
        float mk = (y0[r + B_SZ] > 0) ? 1.0f : 0.0f;
        hp[(size_t)r * K_TAGS + col] = (acc[nt][j] + bv) * mk;
      }
    }
  }
}

// ---------- CRF forward scan + gold score: one wave per batch element ------

__global__ void crf_scan(const float* __restrict__ hp, const int* __restrict__ y0,
                         const float* __restrict__ trans, float* __restrict__ res) {
  int b = blockIdx.x;
  int k = threadIdx.x;           // 0..31
  __shared__ float sc[K_TAGS];
  __shared__ float tmp[32];
  if (k < K_TAGS) sc[k] = (k == 1 /*SOS*/) ? 0.0f : NEGV;

  float trow[K_TAGS];
  if (k < K_TAGS) {
#pragma unroll
    for (int j = 0; j < K_TAGS; ++j) trow[j] = trans[k * K_TAGS + j];
  }
  __syncthreads();

  for (int t = 0; t < T_LEN; ++t) {
    int r = t * B_SZ + b;
    float mt = (y0[r + B_SZ] > 0) ? 1.0f : 0.0f;
    float newk = NEGV;
    if (k < K_TAGS) {
      float ht = hp[(size_t)r * K_TAGS + k];
      float mx = -3.4e38f;
#pragma unroll
      for (int j = 0; j < K_TAGS; ++j) mx = fmaxf(mx, sc[j] + trow[j]);
      float s = 0.0f;
#pragma unroll
      for (int j = 0; j < K_TAGS; ++j) s += __expf(sc[j] + trow[j] - mx);
      newk = mx + __logf(s) + ht;
    }
    __syncthreads();
    if (k < K_TAGS) sc[k] = mt * newk + (1.0f - mt) * sc[k];
    __syncthreads();
  }

  tmp[k] = (k < K_TAGS) ? (sc[k] + trans[2 /*EOS*/ * K_TAGS + k]) : NEGV;
  __syncthreads();
  if (k == 0) {
    float mx = tmp[0];
    for (int i = 1; i < K_TAGS; ++i) mx = fmaxf(mx, tmp[i]);
    float s = 0.0f;
    for (int i = 0; i < K_TAGS; ++i) s += __expf(tmp[i] - mx);
    float Z = mx + __logf(s);

    float g = 0.0f; int lens = 0;
    for (int t = 0; t < T_LEN; ++t) {
      int yt = y0[(t + 1) * B_SZ + b];
      int yp = y0[t * B_SZ + b];
      float mk = (yt > 0) ? 1.0f : 0.0f;
      float emit = hp[(size_t)(t * B_SZ + b) * K_TAGS + yt];
      float tr = trans[yt * K_TAGS + yp];
      g += mk * (emit + tr);
      lens += (yt > 0) ? 1 : 0;
    }
    int last = y0[lens * B_SZ + b];
    g += trans[2 * K_TAGS + last];
    res[b] = Z - g;
  }
}

__global__ void reduce_mean(const float* __restrict__ res, float* __restrict__ out) {
  __shared__ float s[B_SZ];
  int i = threadIdx.x;
  s[i] = res[i];
  __syncthreads();
  if (i == 0) {
    float a = 0.0f;
    for (int j = 0; j < B_SZ; ++j) a += s[j];
    out[0] = a / (float)B_SZ;
  }
}

// ---------------------------- launcher -------------------------------------

extern "C" void kernel_launch(void* const* d_in, const int* in_sizes, int n_in,
                              void* d_out, int out_size, void* d_ws, size_t ws_size,
                              hipStream_t stream) {
  (void)in_sizes; (void)n_in; (void)out_size; (void)ws_size;
  const int*   xw    = (const int*)  d_in[0];
  const int*   y0    = (const int*)  d_in[1];
  const float* emb   = (const float*)d_in[2];
  const float* wihf  = (const float*)d_in[3];
  const float* whhf  = (const float*)d_in[4];
  const float* bf    = (const float*)d_in[5];
  const float* wihb  = (const float*)d_in[6];
  const float* whhb  = (const float*)d_in[7];
  const float* bb    = (const float*)d_in[8];
  const float* wout  = (const float*)d_in[9];
  const float* bout  = (const float*)d_in[10];
  const float* trans = (const float*)d_in[11];

  size_t off = 0;
  char* base = (char*)d_ws;
  auto alloc = [&](size_t bytes) -> void* {
    void* p = base + off;
    off += (bytes + 255) & ~(size_t)255;
    return p;
  };
  const size_t MB_ = (size_t)T_LEN * B_SZ;                 // 16384 rows
  _Float16* x16    = (_Float16*)alloc(MB_ * E_DIM * 2);
  _Float16* wihf16 = (_Float16*)alloc((size_t)G_DIM * E_DIM * 2);
  _Float16* whhf16 = (_Float16*)alloc((size_t)G_DIM * H_DIM * 2);
  _Float16* wihb16 = (_Float16*)alloc((size_t)G_DIM * E_DIM * 2);
  _Float16* whhb16 = (_Float16*)alloc((size_t)G_DIM * H_DIM * 2);
  _Float16* wout16 = (_Float16*)alloc((size_t)32 * 2 * H_DIM * 2);
  _Float16* xgf    = (_Float16*)alloc(MB_ * G_DIM * 2);
  _Float16* xgb    = (_Float16*)alloc(MB_ * G_DIM * 2);
  _Float16* hfb    = (_Float16*)alloc(MB_ * H_DIM * 2);
  _Float16* hbb    = (_Float16*)alloc(MB_ * H_DIM * 2);
  _Float16* hg     = (_Float16*)alloc((size_t)2 * B_SZ * H_DIM * 2);
  unsigned* barcnt = (unsigned*)alloc(128 * 4);
  float*    hproj  = (float*)   alloc(MB_ * K_TAGS * 4);
  float*    res    = (float*)   alloc(B_SZ * 4);

  const int nW = G_DIM * E_DIM;                            // 1048576
  cvt_f16<<<(nW + 255) / 256, 256, 0, stream>>>(wihf, wihf16, nW);
  cvt_f16<<<(nW + 255) / 256, 256, 0, stream>>>(whhf, whhf16, nW);
  cvt_f16<<<(nW + 255) / 256, 256, 0, stream>>>(wihb, wihb16, nW);
  cvt_f16<<<(nW + 255) / 256, 256, 0, stream>>>(whhb, whhb16, nW);
  pad_wout<<<(32 * 1024 + 255) / 256, 256, 0, stream>>>(wout, wout16);
  gather_embed<<<(int)(MB_ * E_DIM / 256), 256, 0, stream>>>(xw, emb, x16);
  init_state<<<(2 * B_SZ * H_DIM + 255) / 256, 256, 0, stream>>>(hg, barcnt);

  // 1024 m-tiles x 32 n-groups = 32768 waves = 4096 blocks of 256
  gemm_xg<<<4096, 256, 0, stream>>>(x16, wihf16, bf, xgf);
  gemm_xg<<<4096, 256, 0, stream>>>(x16, wihb16, bb, xgb);

  lstm_multi<<<2 * NW, 256, 0, stream>>>(xgf, xgb, whhf16, whhb16, hfb, hbb,
                                         hg, barcnt);

  // 1024 waves = 128 blocks of 256
  gemm_proj<<<128, 256, 0, stream>>>(hfb, hbb, wout16, bout, y0, hproj);

  crf_scan<<<B_SZ, 32, 0, stream>>>(hproj, y0, trans, res);
  reduce_mean<<<1, B_SZ, 0, stream>>>(res, (float*)d_out);
}
